// CorrelationModule_25692494365421
// MI455X (gfx1250) — matrix operand order repaired and checked by reading, not compiled, not run on hardware
//
#include <hip/hip_runtime.h>
#include <hip/hip_bf16.h>
#include <math.h>

typedef __attribute__((ext_vector_type(16))) _Float16 v16h;
typedef __attribute__((ext_vector_type(8)))  _Float16 v8h;
typedef __attribute__((ext_vector_type(8)))  float    v8f;

#define LVLS  2
#define BATCH 2
#define FDIM  32
#define HC    64
#define WC    96
#define DD    49          // (2*RADIUS+1)^2
#define NBD   (BATCH*DD)  // 98
#define HWPIX (HC*WC)     // 6144
#define EPSF  1e-5f

// ---------------- workspace layout (bytes) ----------------
static constexpr size_t OFF_BUF0 = 0;                                   // f16 activations (max 64ch @64x96)
static constexpr size_t SZ_BUF0  = (size_t)NBD*HWPIX*64*2;              // 77,070,336
static constexpr size_t OFF_BUF1 = OFF_BUF0 + SZ_BUF0;                  // f16 activations (max 96ch @64x96)
static constexpr size_t SZ_BUF1  = (size_t)NBD*HWPIX*96*2;              // 115,605,504
static constexpr size_t OFF_COST = OFF_BUF1 + SZ_BUF1;                  // f32 cost (98,64,96)
static constexpr size_t SZ_COST  = (size_t)NBD*HWPIX*4;
static constexpr size_t OFF_W0P  = OFF_COST + SZ_COST;                  // packed f16 weights
static constexpr size_t OFF_W1P  = OFF_W0P + (size_t)96*576*2;
static constexpr size_t OFF_W2P  = OFF_W1P + (size_t)128*864*2;
static constexpr size_t OFF_W3P  = OFF_W2P + (size_t)128*1152*2;
static constexpr size_t OFF_B0   = OFF_W3P + (size_t)64*1152*2;         // folded BN biases (f32)
static constexpr size_t OFF_B1   = OFF_B0 + 96*4;
static constexpr size_t OFF_B2   = OFF_B1 + 128*4;
static constexpr size_t OFF_B3   = OFF_B2 + 128*4;
static constexpr size_t OFF_ZERO = OFF_B3 + 64*4;                       // 512B zero page (halo reads)

// ---------------------------------------------------------------------------
// Zero the halo scratch page (re-run every call: deterministic, capture-safe).
// ---------------------------------------------------------------------------
__global__ void zero_pad_kernel(float* __restrict__ z)
{
  z[threadIdx.x] = 0.f;   // 128 threads * 4B = 512B
}

// ---------------------------------------------------------------------------
// Pack conv weights OIHW f32 -> [cout][tap*Cin+cin] f16 with BN scale folded.
// Also emits bias[cout] = b - m*scale.
// ---------------------------------------------------------------------------
__global__ void pack_weights_kernel(const float* __restrict__ w,
                                    const float* __restrict__ g,
                                    const float* __restrict__ bb,
                                    const float* __restrict__ mm,
                                    const float* __restrict__ vv,
                                    _Float16* __restrict__ wp,
                                    float* __restrict__ bias,
                                    int Cout, int Cin, int level)
{
  int idx = blockIdx.x * blockDim.x + threadIdx.x;
  int total = Cout * Cin * 9;
  if (idx < total) {
    int tap = idx % 9;
    int t   = idx / 9;
    int ci  = t % Cin;
    int co  = t / Cin;
    float scale = g[level*Cout + co] * rsqrtf(vv[level*Cout + co] + EPSF);
    wp[(size_t)co*(Cin*9) + tap*Cin + ci] =
        (_Float16)(w[(((size_t)(level*Cout + co))*Cin + ci)*9 + tap] * scale);
  }
  if (idx < Cout) {
    float scale = g[level*Cout + idx] * rsqrtf(vv[level*Cout + idx] + EPSF);
    bias[idx] = bb[level*Cout + idx] - mm[level*Cout + idx] * scale;
  }
}

// ---------------------------------------------------------------------------
// Sampler: builds corr tensor (98, 64, 96, 64) NHWC f16.
// Channels [0,32) = fmap1[level], [32,64) = bilinear sample of fmap2[level]
// at coords/2^level + (dx,dy).  Batch index bd = (b*7 + a)*7 + c where a
// shifts x (first delta axis) and c shifts y (second delta axis).
// ---------------------------------------------------------------------------
__global__ void sampler_kernel(const float* __restrict__ fmap1,
                               const float* __restrict__ fmap2,
                               const float* __restrict__ coords,
                               _Float16* __restrict__ corr, int level)
{
  int idx = blockIdx.x * blockDim.x + threadIdx.x;   // over NBD*HWPIX
  if (idx >= NBD*HWPIX) return;
  int bd  = idx / HWPIX;
  int pix = idx - bd*HWPIX;
  int b = bd / DD, d = bd - b*DD;
  int dxs = d / 7 - 3;   // first delta axis -> x shift
  int dys = d % 7 - 3;   // second delta axis -> y shift
  float inv = (level == 0) ? 1.0f : 0.5f;
  float cx = coords[(b*2 + 0)*HWPIX + pix] * inv + (float)dxs;
  float cy = coords[(b*2 + 1)*HWPIX + pix] * inv + (float)dys;

  const float* F1 = fmap1 + ((size_t)(level*BATCH + b)) * FDIM * HWPIX;
  const float* F2 = fmap2 + ((size_t)(level*BATCH + b)) * FDIM * HWPIX;
  _Float16* op = corr + (size_t)idx * (2*FDIM);

  float x0f = floorf(cx), y0f = floorf(cy);
  float fx = cx - x0f, fy = cy - y0f;
  int ix0 = (int)x0f, iy0 = (int)y0f;
  int ix1 = ix0 + 1,  iy1 = iy0 + 1;
  bool vx0 = (ix0 >= 0) && (ix0 < WC);
  bool vx1 = (ix1 >= 0) && (ix1 < WC);
  bool vy0 = (iy0 >= 0) && (iy0 < HC);
  bool vy1 = (iy1 >= 0) && (iy1 < HC);
  int cx0 = min(max(ix0, 0), WC-1), cx1 = min(max(ix1, 0), WC-1);
  int cy0 = min(max(iy0, 0), HC-1), cy1 = min(max(iy1, 0), HC-1);
  float w00 = (1.f-fx)*(1.f-fy) * ((vx0 && vy0) ? 1.f : 0.f);
  float w01 = fx*(1.f-fy)       * ((vx1 && vy0) ? 1.f : 0.f);
  float w10 = (1.f-fx)*fy       * ((vx0 && vy1) ? 1.f : 0.f);
  float w11 = fx*fy             * ((vx1 && vy1) ? 1.f : 0.f);
  int i00 = cy0*WC + cx0, i01 = cy0*WC + cx1;
  int i10 = cy1*WC + cx0, i11 = cy1*WC + cx1;

  #pragma unroll 4
  for (int c = 0; c < FDIM; ++c) {
    op[c] = (_Float16)F1[(size_t)c*HWPIX + pix];
    const float* Fc = F2 + (size_t)c*HWPIX;
    float s = w00*Fc[i00] + w01*Fc[i01] + w10*Fc[i10] + w11*Fc[i11];
    op[FDIM + c] = (_Float16)s;
  }
}

// ---------------------------------------------------------------------------
// Implicit-GEMM 3x3 conv (pad=1) with fused BN+ReLU epilogue.
// NHWC f16 in/out.  One wave computes a 64(M) x 32(N) output tile with
// 8 accumulators (4 M-subtiles x 2 N-subtiles); K = CIN*9 consumed in
// 32-wide chunks (one 3x3 tap, 32 contiguous channels) via
// v_wmma_f32_16x16x32_f16: 8 WMMAs per 12 b128 loads.
//
// Halo handling: loads are unconditional; out-of-bounds lanes read a zeroed
// scratch page (address select, no EXEC divergence -> no saveexec branches).
//
// A fragment (16x32 f16, per ISA layout): lanes 0-15 hold M=lane,
// K {0..7,16..23}; lanes 16-31 hold M=lane-16, K {8..15,24..31}.
// B fragment (32x16 f16): lane = column N, same K order as A: lanes 0-15
// K {0..7,16..23}; lanes 16-31 K {8..15,24..31} (from packed weight rows).
// C/D (16x16 f32): lanes 0-15: N=lane, M=vgpr; lanes 16-31: N=lane-16, M=vgpr+8.
// ---------------------------------------------------------------------------
template <int CIN, int STRIDE>
__global__ __launch_bounds__(32) void conv3x3_wmma_kernel(
    const _Float16* __restrict__ in, const _Float16* __restrict__ wp,
    const float* __restrict__ bias, const _Float16* __restrict__ zpad,
    _Float16* __restrict__ out,
    int Hin, int Win, int Hout, int Wout, int Cout)
{
  const int lane   = threadIdx.x;
  const int HWo    = Hout * Wout;
  const int tilesN = Cout >> 5;
  const int tN     = blockIdx.x % tilesN;
  const int tM     = blockIdx.x / tilesN;
  const int K      = CIN * 9;

  const int ml  = lane & 15;
  const int kbA = (lane >= 16) ? 8  : 0;   // A-fragment klocal base
  const int kbB = (lane >= 16) ? 8  : 0;   // B-fragment klocal base

  // decode spatial coords of the 4 M-subtiles handled by this lane
  int bd[4], yy[4], xx[4];
  #pragma unroll
  for (int s = 0; s < 4; ++s) {
    int m = tM*64 + s*16 + ml;
    bd[s] = m / HWo;
    int r = m - bd[s]*HWo;
    yy[s] = r / Wout;
    xx[s] = r - yy[s]*Wout;
  }

  const _Float16* w0r = wp + (size_t)(tN*32 + ml) * K + kbB;   // N-subtile 0 row
  const _Float16* w1r = w0r + (size_t)16 * K;                  // N-subtile 1 row
  const _Float16* zp  = zpad + kbA;

  v8f acc0[4] = {};   // N-subtile 0
  v8f acc1[4] = {};   // N-subtile 1

  #pragma unroll
  for (int tap = 0; tap < 9; ++tap) {
    const int dy = tap/3 - 1, dx = tap%3 - 1;

    // per-M-subtile source pointers; halo lanes redirected to zero page
    const _Float16* pa[4];
    #pragma unroll
    for (int s = 0; s < 4; ++s) {
      int yi = yy[s]*STRIDE + dy, xi = xx[s]*STRIDE + dx;
      bool ok = (yi >= 0) & (yi < Hin) & (xi >= 0) & (xi < Win);
      long off = ((long)bd[s]*Hin + yi)*Win + xi;
      pa[s] = ok ? (in + off*CIN + kbA) : zp;
    }

    #pragma unroll
    for (int cb = 0; cb < CIN; cb += 32) {
      v16h A[4];
      #pragma unroll
      for (int s = 0; s < 4; ++s) {
        v8h lo = *(const v8h*)(pa[s] + cb);
        v8h hi = *(const v8h*)(pa[s] + cb + 16);
        A[s] = __builtin_shufflevector(lo, hi, 0,1,2,3,4,5,6,7,8,9,10,11,12,13,14,15);
      }
      const int ko = tap*CIN + cb;
      v8h b0lo = *(const v8h*)(w0r + ko);
      v8h b0hi = *(const v8h*)(w0r + ko + 16);
      v8h b1lo = *(const v8h*)(w1r + ko);
      v8h b1hi = *(const v8h*)(w1r + ko + 16);
      v16h B0 = __builtin_shufflevector(b0lo, b0hi, 0,1,2,3,4,5,6,7,8,9,10,11,12,13,14,15);
      v16h B1 = __builtin_shufflevector(b1lo, b1hi, 0,1,2,3,4,5,6,7,8,9,10,11,12,13,14,15);

      #pragma unroll
      for (int s = 0; s < 4; ++s) {
        acc0[s] = __builtin_amdgcn_wmma_f32_16x16x32_f16(false, A[s], false, B0, (short)0, acc0[s], false, false);
        acc1[s] = __builtin_amdgcn_wmma_f32_16x16x32_f16(false, A[s], false, B1, (short)0, acc1[s], false, false);
      }
    }
  }

  // Epilogue: relu(acc + folded_bias) -> f16 NHWC (out addr = m*Cout + n).
  const int n0 = tN*32 + ml;
  const int n1 = n0 + 16;
  const float bs0 = bias[n0], bs1 = bias[n1];
  const int mr = (lane < 16) ? 0 : 8;
  #pragma unroll
  for (int s = 0; s < 4; ++s) {
    const int mb = tM*64 + s*16 + mr;
    #pragma unroll
    for (int v = 0; v < 8; ++v) {
      float t;
      t = acc0[s][v] + bs0; t = t > 0.f ? t : 0.f; out[(size_t)(mb+v)*Cout + n0] = (_Float16)t;
      t = acc1[s][v] + bs1; t = t > 0.f ? t : 0.f; out[(size_t)(mb+v)*Cout + n1] = (_Float16)t;
    }
  }
}

// ---------------------------------------------------------------------------
// 4x4 stride-2 deconv (lhs_dilation=2, pad=2) 64ch@32x48 -> 32ch@64x96,
// fused BN+ReLU.  Only 4 kernel taps contribute per output pixel (parity).
// k[co][ci][ky][kx] = w4[ci][co][3-ky][3-kx].   (~3% of FLOPs -> VALU.)
// ---------------------------------------------------------------------------
__global__ void deconv_kernel(const _Float16* __restrict__ in,
                              const float* __restrict__ w4,
                              const float* __restrict__ g,
                              const float* __restrict__ bb,
                              const float* __restrict__ mm,
                              const float* __restrict__ vv,
                              _Float16* __restrict__ out, int level)
{
  int idx = blockIdx.x * blockDim.x + threadIdx.x;
  if (idx >= NBD*HWPIX*32) return;
  int co  = idx & 31;
  int pix = idx >> 5;
  int x  = pix % WC;
  int t  = pix / WC;
  int y  = t % HC;
  int bd = t / HC;

  const int py = y & 1, px = x & 1;
  float acc = 0.f;
  for (int a = 0; a < 2; ++a) {
    int ky = py + 2*a;
    int iy = (y - 2 + ky) >> 1;               // (y-2+ky) is even by parity
    if (iy < 0 || iy >= HC/2) continue;
    for (int c2 = 0; c2 < 2; ++c2) {
      int kx = px + 2*c2;
      int ix = (x - 2 + kx) >> 1;
      if (ix < 0 || ix >= WC/2) continue;
      const _Float16* ip = in + (((size_t)bd*(HC/2) + iy)*(WC/2) + ix)*64;
      const float* wpt = w4 + (size_t)level*64*32*16 + co*16 + (3-ky)*4 + (3-kx);
      #pragma unroll 8
      for (int ci = 0; ci < 64; ++ci)
        acc += (float)ip[ci] * wpt[(size_t)ci*512];
    }
  }
  float scale = g[level*32 + co] * rsqrtf(vv[level*32 + co] + EPSF);
  float r = (acc - mm[level*32 + co]) * scale + bb[level*32 + co];
  out[idx] = (_Float16)(r > 0.f ? r : 0.f);
}

// ---------------------------------------------------------------------------
// Final 3x3 conv, 32ch -> 1ch, + bias.  Writes f32 cost (98,64,96).
// ---------------------------------------------------------------------------
__global__ void conv5_kernel(const _Float16* __restrict__ in,
                             const float* __restrict__ w5,
                             const float* __restrict__ b5,
                             float* __restrict__ cost, int level)
{
  int idx = blockIdx.x * blockDim.x + threadIdx.x;
  if (idx >= NBD*HWPIX) return;
  int bd  = idx / HWPIX;
  int pix = idx - bd*HWPIX;
  int y = pix / WC, x = pix - y*WC;
  const float* W = w5 + level*FDIM*9;
  float acc = 0.f;
  #pragma unroll
  for (int tap = 0; tap < 9; ++tap) {
    int dy = tap/3 - 1, dx = tap%3 - 1;
    int yy = y + dy, xx = x + dx;
    if (yy < 0 || yy >= HC || xx < 0 || xx >= WC) continue;
    const _Float16* ip = in + (((size_t)bd*HC + yy)*WC + xx)*FDIM;
    #pragma unroll 8
    for (int c = 0; c < FDIM; ++c)
      acc += (float)ip[c] * W[c*9 + tap];
  }
  cost[idx] = acc + b5[level];
}

// ---------------------------------------------------------------------------
// Displacement-aware projection: out[b, level*49+o, h, w] =
//   sum_d dap_w[level][o][d] * cost[b*49+d, h, w]
// ---------------------------------------------------------------------------
__global__ void dap_kernel(const float* __restrict__ cost,
                           const float* __restrict__ dapw,
                           float* __restrict__ outp, int level)
{
  int idx = blockIdx.x * blockDim.x + threadIdx.x;
  if (idx >= BATCH*DD*HWPIX) return;
  int pix = idx % HWPIX;
  int t   = idx / HWPIX;
  int o = t % DD;
  int b = t / DD;
  const float* Drow = dapw + ((size_t)level*DD + o)*DD;
  float acc = 0.f;
  #pragma unroll 7
  for (int d = 0; d < DD; ++d)
    acc += Drow[d] * cost[((size_t)b*DD + d)*HWPIX + pix];
  outp[(((size_t)b*(2*DD)) + level*DD + o)*HWPIX + pix] = acc;
}

// ---------------------------------------------------------------------------
extern "C" void kernel_launch(void* const* d_in, const int* in_sizes, int n_in,
                              void* d_out, int out_size, void* d_ws, size_t ws_size,
                              hipStream_t stream)
{
  (void)in_sizes; (void)n_in; (void)out_size; (void)ws_size;

  const float* fmap1  = (const float*)d_in[0];
  const float* fmap2  = (const float*)d_in[1];
  const float* coords = (const float*)d_in[2];
  const float* w0  = (const float*)d_in[3];
  const float* g0  = (const float*)d_in[4];
  const float* b0  = (const float*)d_in[5];
  const float* m0  = (const float*)d_in[6];
  const float* v0  = (const float*)d_in[7];
  const float* w1  = (const float*)d_in[8];
  const float* g1  = (const float*)d_in[9];
  const float* b1  = (const float*)d_in[10];
  const float* m1  = (const float*)d_in[11];
  const float* v1  = (const float*)d_in[12];
  const float* w2  = (const float*)d_in[13];
  const float* g2  = (const float*)d_in[14];
  const float* b2  = (const float*)d_in[15];
  const float* m2  = (const float*)d_in[16];
  const float* v2  = (const float*)d_in[17];
  const float* w3  = (const float*)d_in[18];
  const float* g3  = (const float*)d_in[19];
  const float* b3  = (const float*)d_in[20];
  const float* m3  = (const float*)d_in[21];
  const float* v3  = (const float*)d_in[22];
  const float* w4  = (const float*)d_in[23];
  const float* g4  = (const float*)d_in[24];
  const float* b4  = (const float*)d_in[25];
  const float* m4  = (const float*)d_in[26];
  const float* v4  = (const float*)d_in[27];
  const float* w5  = (const float*)d_in[28];
  const float* b5  = (const float*)d_in[29];
  const float* dapw = (const float*)d_in[30];

  char* ws = (char*)d_ws;
  _Float16* buf0 = (_Float16*)(ws + OFF_BUF0);
  _Float16* buf1 = (_Float16*)(ws + OFF_BUF1);
  float*    cost = (float*)   (ws + OFF_COST);
  _Float16* w0p  = (_Float16*)(ws + OFF_W0P);
  _Float16* w1p  = (_Float16*)(ws + OFF_W1P);
  _Float16* w2p  = (_Float16*)(ws + OFF_W2P);
  _Float16* w3p  = (_Float16*)(ws + OFF_W3P);
  float*    b0p  = (float*)(ws + OFF_B0);
  float*    b1p  = (float*)(ws + OFF_B1);
  float*    b2p  = (float*)(ws + OFF_B2);
  float*    b3p  = (float*)(ws + OFF_B3);
  float*    zpad = (float*)(ws + OFF_ZERO);
  float*    outp = (float*)d_out;

  const int PIXBLK = (NBD*HWPIX + 255) / 256;   // 2352 exact

  // zero halo page (512B) once per call
  zero_pad_kernel<<<1, 128, 0, stream>>>(zpad);

  for (int lvl = 0; lvl < LVLS; ++lvl) {
    // pack weights (BN folded)
    pack_weights_kernel<<<(96*64*9   + 255)/256, 256, 0, stream>>>(w0, g0, b0, m0, v0, w0p, b0p,  96,  64, lvl);
    pack_weights_kernel<<<(128*96*9  + 255)/256, 256, 0, stream>>>(w1, g1, b1, m1, v1, w1p, b1p, 128,  96, lvl);
    pack_weights_kernel<<<(128*128*9 + 255)/256, 256, 0, stream>>>(w2, g2, b2, m2, v2, w2p, b2p, 128, 128, lvl);
    pack_weights_kernel<<<(64*128*9  + 255)/256, 256, 0, stream>>>(w3, g3, b3, m3, v3, w3p, b3p,  64, 128, lvl);

    // build corr tensor (f1 broadcast ++ bilinear f2 samples), NHWC f16
    sampler_kernel<<<PIXBLK, 256, 0, stream>>>(fmap1, fmap2, coords, buf0, lvl);

    // matching net convs via WMMA implicit GEMM (64M x 32N per wave)
    //   L0: 64 -> 96 @64x96   M=602112 -> 9408 M-tiles * 3 N-tiles
    conv3x3_wmma_kernel<64, 1><<<9408*3, 32, 0, stream>>>(buf0, w0p, b0p, (const _Float16*)zpad, buf1, 64, 96, 64, 96,  96);
    //   L1: 96 -> 128 stride2 @32x48   M=150528 -> 2352 * 4
    conv3x3_wmma_kernel<96, 2><<<2352*4, 32, 0, stream>>>(buf1, w1p, b1p, (const _Float16*)zpad, buf0, 64, 96, 32, 48, 128);
    //   L2: 128 -> 128 @32x48
    conv3x3_wmma_kernel<128,1><<<2352*4, 32, 0, stream>>>(buf0, w2p, b2p, (const _Float16*)zpad, buf1, 32, 48, 32, 48, 128);
    //   L3: 128 -> 64 @32x48
    conv3x3_wmma_kernel<128,1><<<2352*2, 32, 0, stream>>>(buf1, w3p, b3p, (const _Float16*)zpad, buf0, 32, 48, 32, 48,  64);

    // L4: deconv 64 -> 32, upsample to 64x96, BN+ReLU
    deconv_kernel<<<(NBD*HWPIX*32 + 255)/256, 256, 0, stream>>>(buf0, w4, g4, b4, m4, v4, buf1, lvl);

    // L5: 32 -> 1 head, f32 cost
    conv5_kernel<<<PIXBLK, 256, 0, stream>>>(buf1, w5, b5, cost, lvl);

    // DAP 49x49 projection into output channels [lvl*49, lvl*49+49)
    dap_kernel<<<(BATCH*DD*HWPIX + 255)/256, 256, 0, stream>>>(cost, dapw, outp, lvl);
  }
}
